// MultiUniverseToposAttention_40561671143614
// MI455X (gfx1250) — compile-verified, matrix operand order checked by reading, not run on hardware
//
#include <hip/hip_runtime.h>

// ---------------------------------------------------------------------------
// MultiUniverseToposAttention for MI455X (gfx1250, wave32)
//   D_MODEL=256, H=8, du=32, B=1, S=1024, TEMP=5
//   truth[i,j] = 1 - (1/32) * sum_d relu(Q[i,d]-K[j,d])   (NOT a matmul -> VALU)
//   projections & output GEMM -> v_wmma_f32_16x16x32_f16
//   logits = 5*truth <= 5  => fixed-max softmax (no running max needed)
//   K/V staging: global_load_async_to_lds_b128 (ASYNCcnt) double buffered.
// ---------------------------------------------------------------------------

typedef _Float16 v16h __attribute__((ext_vector_type(16)));
typedef _Float16 h8   __attribute__((ext_vector_type(8)));
typedef float    v8f  __attribute__((ext_vector_type(8)));

union AV { v16h v; h8 h[2]; };

#define S_LEN   1024
#define D_MODEL 256
#define NHEAD   8
#define DU      32
#define TEMP_F  5.0f

#if __has_builtin(__builtin_amdgcn_global_load_async_to_lds_b128) && \
    __has_builtin(__builtin_amdgcn_s_wait_asynccnt)
#define HAVE_ASYNC_LDS 1
// Exact parameter types per the compiler diagnostic:
//   param0: int __attribute__((vector_size(16))) AS(1)*   (non-const)
//   param1: int __attribute__((vector_size(16))) AS(3)*
typedef int v4i_t __attribute__((vector_size(16)));
typedef __attribute__((address_space(1))) v4i_t* as1_v4i;
typedef __attribute__((address_space(3))) v4i_t* as3_v4i;
#else
#define HAVE_ASYNC_LDS 0
#endif

// ---------------------------------------------------------------------------
// Kernel 1: pack x -> f16, W -> f16 transposed (B-operand friendly), biases
// grid.x = 1024 (x rows) + 1024 (4*256 W^T rows) + 1 (biases), block = 256
// ---------------------------------------------------------------------------
__global__ __launch_bounds__(256) void pack_kernel(
    const float* __restrict__ x,
    const float* __restrict__ Wq, const float* __restrict__ Wk,
    const float* __restrict__ Wv, const float* __restrict__ Wo,
    const float* __restrict__ bq, const float* __restrict__ bk,
    const float* __restrict__ bv, const float* __restrict__ bo,
    _Float16* __restrict__ xh, _Float16* __restrict__ WT_all,
    float* __restrict__ bias_all)
{
    const int c = threadIdx.x;
    const int r = blockIdx.x;
    if (r < S_LEN) {
        xh[r * D_MODEL + c] = (_Float16)x[r * D_MODEL + c];
    } else if (r < 2 * S_LEN) {
        const int wr = r - S_LEN;
        const int w  = wr >> 8;        // which weight matrix
        const int n  = wr & 255;       // output-channel row of W^T
        const float* W = (w == 0) ? Wq : (w == 1) ? Wk : (w == 2) ? Wv : Wo;
        WT_all[(w * D_MODEL + n) * D_MODEL + c] = (_Float16)W[c * D_MODEL + n];
    } else {
        bias_all[0 * D_MODEL + c] = bq[c];
        bias_all[1 * D_MODEL + c] = bk[c];
        bias_all[2 * D_MODEL + c] = bv[c];
        bias_all[3 * D_MODEL + c] = bo[c];
    }
}

// ---------------------------------------------------------------------------
// Kernel 2: WMMA GEMM  C[16x16] = A[16x256] * B[256x16] (+bias, +activation)
//   mode 0: Q = sigmoid(x Wq + bq) -> Qf [H][S][32] f32
//   mode 1: K = sigmoid(x Wk + bk) -> Kf [H][S][32] f32
//   mode 2: V = x Wv + bv          -> Vf [H][S][32] f32
//   mode 3: out = ctx Wo + bo      -> d_out [S][256] f32
// block = (32,8): 8 waves, one 16x16 tile each.  grid = (64, 2, nz)
// A-operand f16 layout: lane l: row = l&15, halves j -> k = j + 8*(j>=8) + 8*(l>>4)
// B-operand uses W^T so each lane reads a contiguous row (same index math).
// D layout: lane l: n = l&15, vgpr r: m = r + 8*(l>>4)
// ---------------------------------------------------------------------------
__global__ __launch_bounds__(256) void wmma_gemm_kernel(
    const _Float16* __restrict__ xh, const _Float16* __restrict__ ctxh,
    const _Float16* __restrict__ WT_all, const float* __restrict__ bias_all,
    float* __restrict__ Qf, float* __restrict__ Kf, float* __restrict__ Vf,
    float* __restrict__ out, int modeArg)
{
    const int mode = (modeArg < 0) ? (int)blockIdx.z : modeArg;
    const int lane = threadIdx.x;
    const int l15  = lane & 15;
    const int hi   = lane >> 4;
    const int mt   = blockIdx.x;
    const int nt   = blockIdx.y * 8 + threadIdx.y;

    const _Float16* A  = (mode == 3) ? ctxh : xh;
    const _Float16* BT = WT_all + mode * (D_MODEL * D_MODEL);

    const _Float16* arow = A  + (mt * 16 + l15) * D_MODEL;
    const _Float16* brow = BT + (nt * 16 + l15) * D_MODEL;

    v8f c = {};
#pragma unroll
    for (int k0 = 0; k0 < D_MODEL; k0 += 32) {
        AV a, b;
        a.h[0] = *(const h8*)(arow + k0 +      8 * hi);
        a.h[1] = *(const h8*)(arow + k0 + 16 + 8 * hi);
        b.h[0] = *(const h8*)(brow + k0 +      8 * hi);
        b.h[1] = *(const h8*)(brow + k0 + 16 + 8 * hi);
        c = __builtin_amdgcn_wmma_f32_16x16x32_f16(
                false, a.v, false, b.v, (short)0, c, false, false);
    }

    const int nglob = nt * 16 + l15;
    const float bias = bias_all[mode * D_MODEL + nglob];
    const int h = nglob >> 5;
    const int d = nglob & 31;

#pragma unroll
    for (int r = 0; r < 8; ++r) {
        const int m = r + hi * 8;
        const int s = mt * 16 + m;
        float v = c[r] + bias;
        if (mode <= 1) {
            v = 1.0f / (1.0f + __expf(-v));      // sigmoid (fuzzy truth)
            float* dst = (mode == 0) ? Qf : Kf;
            dst[(h * S_LEN + s) * DU + d] = v;
        } else if (mode == 2) {
            Vf[(h * S_LEN + s) * DU + d] = v;
        } else {
            out[s * D_MODEL + nglob] = v;
        }
    }
}

// ---------------------------------------------------------------------------
// K/V block staging into LDS: async (ASYNCcnt, VGPR-bypassing) when available.
// 128 threads move 64x32 f32 K + 64x32 f32 V (8 KB + 8 KB) per block.
// ---------------------------------------------------------------------------
__device__ __forceinline__ void stage_kv(const float* kg, const float* vg,
                                         float* sk, float* sv, int tid)
{
#if HAVE_ASYNC_LDS
    float* kgm = const_cast<float*>(kg);
    float* vgm = const_cast<float*>(vg);
#pragma unroll
    for (int i = 0; i < 4; ++i) {
        const int o = (tid + i * 128) * 4;     // float index, 16B aligned
        __builtin_amdgcn_global_load_async_to_lds_b128(
            (as1_v4i)(kgm + o), (as3_v4i)(sk + o), 0, 0);
        __builtin_amdgcn_global_load_async_to_lds_b128(
            (as1_v4i)(vgm + o), (as3_v4i)(sv + o), 0, 0);
    }
#else
#pragma unroll
    for (int i = 0; i < 4; ++i) {
        ((float4*)sk)[tid + i * 128] = ((const float4*)kg)[tid + i * 128];
        ((float4*)sv)[tid + i * 128] = ((const float4*)vg)[tid + i * 128];
    }
#endif
}

__device__ __forceinline__ void wait_async_lds()
{
#if HAVE_ASYNC_LDS
    __builtin_amdgcn_s_wait_asynccnt(0);       // own async->LDS ops complete
#endif
}

// ---------------------------------------------------------------------------
// Kernel 3: fused causal fuzzy attention (flash-style, fixed-max softmax)
// block = (32,4): 4 waves x 32 query rows; grid = (S/128, H)
// Double-buffered K/V in LDS; stage of block jb+1 overlaps compute of jb.
// truth<=1 => logit<=TEMP => p=exp(TEMP*truth - TEMP) needs no running max.
// ---------------------------------------------------------------------------
__global__ __launch_bounds__(128) void attn_kernel(
    const float* __restrict__ Qf, const float* __restrict__ Kf,
    const float* __restrict__ Vf, _Float16* __restrict__ ctxh)
{
    __shared__ float Klds[2][64 * DU];
    __shared__ float Vlds[2][64 * DU];

    const int head = blockIdx.y;
    const int lane = threadIdx.x;
    const int wv   = threadIdx.y;
    const int tid  = wv * 32 + lane;
    const int q    = blockIdx.x * 128 + wv * 32 + lane;   // this lane's query row

    float4 qv[8], ctx[8];
    const float4* qrow = (const float4*)(Qf + (head * S_LEN + q) * DU);
#pragma unroll
    for (int i = 0; i < 8; ++i) { qv[i] = qrow[i]; ctx[i] = make_float4(0.f, 0.f, 0.f, 0.f); }
    float lsum = 0.0f;

    const int wave_qmax = blockIdx.x * 128 + wv * 32 + 31;  // wave-uniform
    const int blk_qmax  = blockIdx.x * 128 + 127;           // block-uniform
    const int nb        = (blk_qmax >> 6) + 1;              // # 64-key blocks

    const float* kbase = Kf + head * S_LEN * DU;
    const float* vbase = Vf + head * S_LEN * DU;

    stage_kv(kbase, vbase, Klds[0], Vlds[0], tid);          // prologue: block 0

    for (int jb = 0; jb < nb; ++jb) {
        wait_async_lds();        // my stage of buf[jb&1] is done (issued last iter)
        __syncthreads();         // everyone's stage done + everyone off buf[(jb+1)&1]
        if (jb + 1 < nb) {
            stage_kv(kbase + (jb + 1) * 64 * DU, vbase + (jb + 1) * 64 * DU,
                     Klds[(jb + 1) & 1], Vlds[(jb + 1) & 1], tid);
        }
        if (jb + 2 < nb) {       // warm L2 for the block after next (CDNA5 prefetch)
            __builtin_prefetch(kbase + (jb + 2) * 64 * DU + tid * 16, 0, 1);
            __builtin_prefetch(vbase + (jb + 2) * 64 * DU + tid * 16, 0, 1);
        }

        const float* Kb = Klds[jb & 1];
        const float* Vb = Vlds[jb & 1];
        int jmax = wave_qmax - jb * 64;          // wave-uniform trip count
        jmax = (jmax > 63) ? 63 : jmax;

        for (int j = 0; j <= jmax; ++j) {
            const float4* Kr = (const float4*)(Kb + j * DU);  // broadcast ds reads
            float f0 = 0.f, f1 = 0.f, f2 = 0.f, f3 = 0.f;
#pragma unroll
            for (int d4 = 0; d4 < 8; ++d4) {
                const float4 kv = Kr[d4];
                f0 += fmaxf(qv[d4].x - kv.x, 0.f);
                f1 += fmaxf(qv[d4].y - kv.y, 0.f);
                f2 += fmaxf(qv[d4].z - kv.z, 0.f);
                f3 += fmaxf(qv[d4].w - kv.w, 0.f);
            }
            const float truth = 1.0f - (f0 + f1 + f2 + f3) * (1.0f / 32.0f);
            const int   jg    = jb * 64 + j;
            const float p = (jg <= q) ? __expf(fmaf(TEMP_F, truth, -TEMP_F)) : 0.0f;
            lsum += p;
            const float4* Vr = (const float4*)(Vb + j * DU);
#pragma unroll
            for (int d4 = 0; d4 < 8; ++d4) {
                const float4 vv = Vr[d4];
                ctx[d4].x = fmaf(p, vv.x, ctx[d4].x);
                ctx[d4].y = fmaf(p, vv.y, ctx[d4].y);
                ctx[d4].z = fmaf(p, vv.z, ctx[d4].z);
                ctx[d4].w = fmaf(p, vv.w, ctx[d4].w);
            }
        }
    }

    const float inv = 1.0f / lsum;               // lsum >= e^-5 (j=0 always valid)
    _Float16* orow = ctxh + q * D_MODEL + head * DU;
#pragma unroll
    for (int i = 0; i < 8; ++i) {
        orow[i * 4 + 0] = (_Float16)(ctx[i].x * inv);
        orow[i * 4 + 1] = (_Float16)(ctx[i].y * inv);
        orow[i * 4 + 2] = (_Float16)(ctx[i].z * inv);
        orow[i * 4 + 3] = (_Float16)(ctx[i].w * inv);
    }
}

// ---------------------------------------------------------------------------
// Host launcher.  Workspace layout (bytes):
//   xh       f16 [1024][256]            @ 0         (512 KB)
//   WT_all   f16 [4][256][256]          @ 0x080000  (512 KB)
//   bias_all f32 [4][256]               @ 0x100000  (4 KB)
//   Qf       f32 [8][1024][32]          @ 0x101000  (1 MB)
//   Kf       f32 [8][1024][32]          @ 0x201000  (1 MB)
//   Vf       f32 [8][1024][32]          @ 0x301000  (1 MB)
//   ctxh     f16 [1024][256]            @ 0x401000  (512 KB)   total ~4.5 MB
// ---------------------------------------------------------------------------
extern "C" void kernel_launch(void* const* d_in, const int* in_sizes, int n_in,
                              void* d_out, int out_size, void* d_ws, size_t ws_size,
                              hipStream_t stream)
{
    const float* x  = (const float*)d_in[0];
    const float* Wq = (const float*)d_in[1];
    const float* bq = (const float*)d_in[2];
    const float* Wk = (const float*)d_in[3];
    const float* bk = (const float*)d_in[4];
    const float* Wv = (const float*)d_in[5];
    const float* bv = (const float*)d_in[6];
    const float* Wo = (const float*)d_in[7];
    const float* bo = (const float*)d_in[8];

    char* ws = (char*)d_ws;
    _Float16* xh       = (_Float16*)(ws + 0x000000);
    _Float16* WT_all   = (_Float16*)(ws + 0x080000);
    float*    bias_all = (float*)   (ws + 0x100000);
    float*    Qf       = (float*)   (ws + 0x101000);
    float*    Kf       = (float*)   (ws + 0x201000);
    float*    Vf       = (float*)   (ws + 0x301000);
    _Float16* ctxh     = (_Float16*)(ws + 0x401000);
    float*    out      = (float*)d_out;

    // 1) pack f16 operands + transposed weights + biases
    pack_kernel<<<dim3(2 * S_LEN + 1), dim3(256), 0, stream>>>(
        x, Wq, Wk, Wv, Wo, bq, bk, bv, bo, xh, WT_all, bias_all);

    // 2) Q/K/V projections via WMMA (grid.z selects matrix)
    wmma_gemm_kernel<<<dim3(64, 2, 3), dim3(32, 8), 0, stream>>>(
        xh, ctxh, WT_all, bias_all, Qf, Kf, Vf, out, -1);

    // 3) fused causal fuzzy-implication attention
    attn_kernel<<<dim3(S_LEN / 128, NHEAD), dim3(32, 4), 0, stream>>>(
        Qf, Kf, Vf, ctxh);

    // 4) output projection via WMMA
    wmma_gemm_kernel<<<dim3(64, 2, 1), dim3(32, 8), 0, stream>>>(
        xh, ctxh, WT_all, bias_all, Qf, Kf, Vf, out, 3);
}
